// GNNModel_19670950216561
// MI455X (gfx1250) — matrix operand order repaired and checked by reading, not compiled, not run on hardware
//
#include <hip/hip_runtime.h>
#include <hip/hip_bf16.h>
#include <math.h>

#define HEADS 4
#define CH    32
#define DIM   128          // HEADS*CH
#define NGRAPH 64
#define EPS_SM 1e-16f
#define EPS_BN 1e-5f

typedef __attribute__((ext_vector_type(16))) __bf16 vbf16x16;
typedef __attribute__((ext_vector_type(8)))  float  vf32x8;

// ---------------- ordered-int float max helpers (handles negatives) --------
__device__ __forceinline__ int f2ord(float f) {
    int i = __float_as_int(f);
    return (i >= 0) ? i : (i ^ 0x7FFFFFFF);
}
__device__ __forceinline__ float ord2f(int i) {
    int j = (i >= 0) ? i : (i ^ 0x7FFFFFFF);
    return __int_as_float(j);
}

// ---------------------------------------------------------------------------
// GEMM: H[N x 128] = A[N x 128] (f32) @ W[128 x 128] (f32), via bf16 WMMA,
// f32 accumulate. One block = 64 rows x 128 cols; 8 waves, each wave owns one
// 16-wide column tile and 4 M-tiles (16 WMMAs per wave). W is staged once per
// block (transposed [col][k]) and each B fragment is reused by 4 WMMAs.
// ---------------------------------------------------------------------------
__global__ __launch_bounds__(256) void gemm_bf16_wmma(
    const float* __restrict__ A, const float* __restrict__ W,
    float* __restrict__ H, int nRows)
{
    __shared__ __bf16 sA[64 * DIM];    // [row][k]            16 KB
    __shared__ __bf16 sW[DIM * DIM];   // [col][k] (W^T)      32 KB

    const int t = threadIdx.x;
    const int rowBase = blockIdx.x * 64;

    // stage A tile (coalesced reads; zero-fill past end so WMMA EXEC stays full)
    for (int i = t; i < 64 * DIM; i += 256) {
        int r = rowBase + (i >> 7);
        sA[i] = (r < nRows) ? (__bf16)A[(size_t)r * DIM + (i & 127)]
                            : (__bf16)0.0f;
    }
    // stage W transposed: W[k*128+col] -> sW[col*128+k]
    for (int i = t; i < DIM * DIM; i += 256) {
        int k = i >> 7, col = i & 127;
        sW[col * DIM + k] = (__bf16)W[i];
    }
    __syncthreads();

    const int wave = t >> 5;          // 0..7 -> column tile
    const int lane = t & 31;
    const int nsub = lane & 15;
    const int hi   = lane >> 4;       // 0 or 1
    const int colBase = wave * 16;

    vf32x8 acc[4] = {};
#pragma unroll
    for (int kt = 0; kt < 4; ++kt) {
        const int kb = kt * 32;

        // B fragment (shared across the 4 M-tiles): 16-bit B 32x16 layout.
        // lanes 0-15 hold K=kb+{0..7},kb+16+{0..7} of column N=lane; lanes 16-31 K=kb+8+{0..7},kb+24+{0..7}
        vbf16x16 bF;
        const __bf16* br = &sW[(colBase + nsub) * DIM + kb + hi * 8];
#pragma unroll
        for (int i = 0; i < 8; ++i) { bF[i] = br[i]; bF[8 + i] = br[16 + i]; }

#pragma unroll
        for (int m = 0; m < 4; ++m) {
            // A fragment: 16-bit A 16x32 layout.
            // lanes 0-15 : K = kb+{0..7} (VGPR0-3), kb+16+{0..7} (VGPR4-7)
            // lanes 16-31: K = kb+8+{0..7},         kb+24+{0..7}
            vbf16x16 aF;
            const __bf16* ar = &sA[(m * 16 + nsub) * DIM + kb + hi * 8];
#pragma unroll
            for (int i = 0; i < 8; ++i) aF[i] = ar[i];
#pragma unroll
            for (int i = 0; i < 8; ++i) aF[8 + i] = ar[16 + i];

            acc[m] = __builtin_amdgcn_wmma_f32_16x16x32_bf16(
                         false, aF, false, bF, (short)0, acc[m], false, false);
        }
    }

    // C/D layout: VGPR r -> M = r (+8 for lanes 16-31), N = lane&15
#pragma unroll
    for (int m = 0; m < 4; ++m) {
        const int mBase = rowBase + m * 16;
        if (mBase < nRows) {                     // wave-uniform guard
#pragma unroll
            for (int r = 0; r < 8; ++r) {
                int mm = r + hi * 8;
                H[(size_t)(mBase + mm) * DIM + colBase + nsub] = acc[m][r];
            }
        }
    }
}

// ---------------- attention coefficients e_src / e_dst --------------------
__global__ void attn_coef(const float* __restrict__ H,
                          const float* __restrict__ a_src,
                          const float* __restrict__ a_dst,
                          float* __restrict__ es, float* __restrict__ ed, int N)
{
    int i = blockIdx.x * blockDim.x + threadIdx.x;          // (node, head)
    int stride = gridDim.x * blockDim.x;
    for (; i < N * HEADS; i += stride) {
        int n = i >> 2, h = i & 3;
        const float* hp = H + (size_t)n * DIM + h * CH;
        const float* ap = a_src + h * CH;
        const float* dp = a_dst + h * CH;
        float s0 = 0.f, s1 = 0.f;
#pragma unroll
        for (int c = 0; c < CH; ++c) { s0 += hp[c] * ap[c]; s1 += hp[c] * dp[c]; }
        es[i] = s0; ed[i] = s1;
    }
}

// ---------------- per-layer init -------------------------------------------
__global__ void init_layer(float* __restrict__ acc, int* __restrict__ mo,
                           float* __restrict__ den, float* __restrict__ bns, int N)
{
    int i = blockIdx.x * blockDim.x + threadIdx.x;
    int stride = gridDim.x * blockDim.x;
    const int negInf = f2ord(-INFINITY);
    for (int j = i; j < N * DIM; j += stride) acc[j] = 0.f;
    for (int j = i; j < N * HEADS; j += stride) { mo[j] = negInf; den[j] = 0.f; }
    for (int j = i; j < 2 * DIM; j += stride) bns[j] = 0.f;
}

// ---------------- edge pass 1: segment max ---------------------------------
__global__ void edge_max(const int* __restrict__ src, const int* __restrict__ dst,
                         const float* __restrict__ es, const float* __restrict__ ed,
                         int* __restrict__ mo, int E, int N)
{
    int i = blockIdx.x * blockDim.x + threadIdx.x;
    int stride = gridDim.x * blockDim.x;
    const int ET = E + N;
    for (; i < ET; i += stride) {
        int s, d;
        if (i < E) { s = src[i]; d = dst[i]; } else { s = d = i - E; }
#pragma unroll
        for (int h = 0; h < HEADS; ++h) {
            float e = es[s * HEADS + h] + ed[d * HEADS + h];
            e = (e > 0.f) ? e : 0.2f * e;                   // leaky_relu 0.2
            atomicMax(&mo[d * HEADS + h], f2ord(e));
        }
    }
}

// ---------------- edge pass 2: exp + denom + weighted aggregate ------------
// wave-per-edge; lane covers 4 contiguous channels. Softmax division is
// deferred to node_finalize (denominator constant per destination). The next
// grid-stride iteration's source row is prefetched (global_prefetch_b8) to
// hide the data-dependent gather latency behind this edge's atomics.
__global__ __launch_bounds__(256) void edge_agg(
    const int* __restrict__ src, const int* __restrict__ dst,
    const float* __restrict__ es, const float* __restrict__ ed,
    const int* __restrict__ mo, const float* __restrict__ H,
    float* __restrict__ den, float* __restrict__ acc, int E, int N)
{
    const int lane   = threadIdx.x & 31;
    const int wave   = (blockIdx.x * blockDim.x + threadIdx.x) >> 5;
    const int nWaves = (gridDim.x * blockDim.x) >> 5;
    const int ET = E + N;
    const int c0 = lane * 4;
    const int head = lane >> 3;          // 32 ch per head / 4 per lane

    for (int e = wave; e < ET; e += nWaves) {
        int s, d;
        if (e < E) { s = src[e]; d = dst[e]; } else { s = d = e - E; }

        // prefetch next iteration's source feature row
        int en = e + nWaves;
        if (en < ET) {
            int sp = (en < E) ? src[en] : (en - E);
            __builtin_prefetch(H + (size_t)sp * DIM + c0, 0, 0);
        }

        float ev = es[s * HEADS + head] + ed[d * HEADS + head];
        ev = (ev > 0.f) ? ev : 0.2f * ev;
        float ex = __expf(ev - ord2f(mo[d * HEADS + head]));
        if ((lane & 7) == 0) atomicAdd(&den[d * HEADS + head], ex);

        const float4 hv = *(const float4*)(H + (size_t)s * DIM + c0);
        float* ap = acc + (size_t)d * DIM + c0;
        atomicAdd(ap + 0, hv.x * ex);
        atomicAdd(ap + 1, hv.y * ex);
        atomicAdd(ap + 2, hv.z * ex);
        atomicAdd(ap + 3, hv.w * ex);
    }
}

// ---------------- node finalize: /denom, +bias, *aw; fused BN stats --------
__global__ __launch_bounds__(256) void node_finalize(
    float* __restrict__ acc, const float* __restrict__ den,
    const float* __restrict__ bias, const int* __restrict__ ntype,
    float* __restrict__ bns, int N)
{
    __shared__ float ss[DIM], sq[DIM];
    if (threadIdx.x < DIM) { ss[threadIdx.x] = 0.f; sq[threadIdx.x] = 0.f; }
    __syncthreads();

    int i = blockIdx.x * blockDim.x + threadIdx.x;
    int stride = gridDim.x * blockDim.x;
    for (int j = i; j < N * DIM; j += stride) {
        int n = j >> 7, c = j & 127;
        float v = acc[j] / (den[n * HEADS + (c >> 5)] + EPS_SM) + bias[c];
        v *= (ntype[n] == 0) ? 1.5f : 1.0f;
        acc[j] = v;
        atomicAdd(&ss[c], v);
        atomicAdd(&sq[c], v * v);
    }
    __syncthreads();
    if (threadIdx.x < DIM) {
        atomicAdd(&bns[threadIdx.x],       ss[threadIdx.x]);
        atomicAdd(&bns[DIM + threadIdx.x], sq[threadIdx.x]);
    }
}

// ---------------- BN (training stats) + ReLU -------------------------------
__global__ void bn_relu(const float* __restrict__ acc, const float* __restrict__ bns,
                        const float* __restrict__ g, const float* __restrict__ be,
                        float* __restrict__ out, int N)
{
    const float invN = 1.0f / (float)N;
    int i = blockIdx.x * blockDim.x + threadIdx.x;
    int stride = gridDim.x * blockDim.x;
    for (int j = i; j < N * DIM; j += stride) {
        int c = j & 127;
        float mu  = bns[c] * invN;
        float var = bns[DIM + c] * invN - mu * mu;
        float y = (acc[j] - mu) * rsqrtf(var + EPS_BN) * g[c] + be[c];
        out[j] = fmaxf(y, 0.f);
    }
}

// ---------------- pooling --------------------------------------------------
__global__ void pool_init(float* __restrict__ psum, float* __restrict__ pcnt)
{
    int i = blockIdx.x * blockDim.x + threadIdx.x;
    int stride = gridDim.x * blockDim.x;
    for (int j = i; j < NGRAPH * DIM; j += stride) psum[j] = 0.f;
    for (int j = i; j < NGRAPH; j += stride) pcnt[j] = 0.f;
}

__global__ void pool_sum(const float* __restrict__ h, const int* __restrict__ batch,
                         float* __restrict__ psum, float* __restrict__ pcnt, int N)
{
    int i = blockIdx.x * blockDim.x + threadIdx.x;
    int stride = gridDim.x * blockDim.x;
    for (int j = i; j < N * DIM; j += stride) {
        int n = j >> 7, c = j & 127;
        int g = batch[n];
        atomicAdd(&psum[g * DIM + c], h[j]);
        if (c == 0) atomicAdd(&pcnt[g], 1.0f);
    }
}

// ---------------- MLP head: relu(pooled@fc1+b1)@fc2+b2 ---------------------
__global__ __launch_bounds__(32) void head_mlp(
    const float* __restrict__ psum, const float* __restrict__ pcnt,
    const float* __restrict__ w1, const float* __restrict__ b1,
    const float* __restrict__ w2, const float* __restrict__ b2,
    float* __restrict__ out)
{
    __shared__ float pooled[DIM];
    __shared__ float red[CH];
    const int g = blockIdx.x, t = threadIdx.x;   // 32 threads
    const float inv = 1.0f / fmaxf(pcnt[g], 1.0f);
    for (int c = t; c < DIM; c += 32) pooled[c] = psum[g * DIM + c] * inv;
    __syncthreads();

    float a = b1[t];
    for (int c = 0; c < DIM; ++c) a += pooled[c] * w1[c * CH + t];
    red[t] = fmaxf(a, 0.f) * w2[t];
    __syncthreads();
    if (t == 0) {
        float s = b2[0];
#pragma unroll
        for (int j = 0; j < CH; ++j) s += red[j];
        out[g] = s;
    }
}

// ===========================================================================
extern "C" void kernel_launch(void* const* d_in, const int* in_sizes, int n_in,
                              void* d_out, int out_size, void* d_ws, size_t ws_size,
                              hipStream_t stream)
{
    const float* x     = (const float*)d_in[0];
    const int*   ei    = (const int*)d_in[1];
    const int*   ntype = (const int*)d_in[2];
    const int*   batch = (const int*)d_in[3];

    const int N = in_sizes[0] / DIM;       // 50000
    const int E = in_sizes[1] / 2;         // 800000
    const int* src = ei;
    const int* dst = ei + E;

    struct LayerW { const float *W, *as, *ad, *b, *g, *be; };
    LayerW L[3];
    for (int l = 0; l < 3; ++l) {
        int o = 4 + l * 6;
        L[l] = { (const float*)d_in[o+0], (const float*)d_in[o+1],
                 (const float*)d_in[o+2], (const float*)d_in[o+3],
                 (const float*)d_in[o+4], (const float*)d_in[o+5] };
    }
    const float* fc1_w = (const float*)d_in[22];
    const float* fc1_b = (const float*)d_in[23];
    const float* fc2_w = (const float*)d_in[24];
    const float* fc2_b = (const float*)d_in[25];
    float* out = (float*)d_out;

    // ---- workspace carve-up (all 4-byte typed) ----
    float* ws   = (float*)d_ws;
    float* h    = ws;                 ws += (size_t)N * DIM;   // GEMM output
    float* acc  = ws;                 ws += (size_t)N * DIM;   // aggregation
    float* hA   = ws;                 ws += (size_t)N * DIM;   // layer activations
    float* es   = ws;                 ws += (size_t)N * HEADS;
    float* ed   = ws;                 ws += (size_t)N * HEADS;
    int*   mo   = (int*)ws;           ws += (size_t)N * HEADS;
    float* den  = ws;                 ws += (size_t)N * HEADS;
    float* bns  = ws;                 ws += 2 * DIM;
    float* psum = ws;                 ws += NGRAPH * DIM;
    float* pcnt = ws;                 ws += NGRAPH;

    const int T = 256;
    const int gGemm = (N + 63) / 64;                  // 64 rows per block
    const int gBig  = 2048;                           // grid-stride kernels
    const int gAttn = (N * HEADS + T - 1) / T;

    const float* in = x;
    for (int l = 0; l < 3; ++l) {
        gemm_bf16_wmma<<<gGemm, T, 0, stream>>>(in, L[l].W, h, N);
        attn_coef<<<gAttn, T, 0, stream>>>(h, L[l].as, L[l].ad, es, ed, N);
        init_layer<<<gBig, T, 0, stream>>>(acc, mo, den, bns, N);
        edge_max<<<gBig, T, 0, stream>>>(src, dst, es, ed, mo, E, N);
        edge_agg<<<gBig, T, 0, stream>>>(src, dst, es, ed, mo, h, den, acc, E, N);
        node_finalize<<<gBig, T, 0, stream>>>(acc, den, L[l].b, ntype, bns, N);
        bn_relu<<<gBig, T, 0, stream>>>(acc, bns, L[l].g, L[l].be, hA, N);
        in = hA;
    }

    pool_init<<<34, T, 0, stream>>>(psum, pcnt);
    pool_sum<<<gBig, T, 0, stream>>>(hA, batch, psum, pcnt, N);
    head_mlp<<<NGRAPH, 32, 0, stream>>>(psum, pcnt, fc1_w, fc1_b, fc2_w, fc2_b, out);

    (void)n_in; (void)out_size; (void)ws_size;
}
